// SwinTransformer_CDPNET3_36301063586406
// MI455X (gfx1250) — matrix operand order repaired and checked by reading, not compiled, not run on hardware
//
#include <hip/hip_runtime.h>
#include <hip/hip_bf16.h>

// ---------------- problem constants ----------------
constexpr int CDIM  = 256;
constexpr int HEADS = 8;
constexpr int HD    = 32;      // head dim
constexpr int NTOK  = 49;      // tokens per window
constexpr int NPAD  = 64;      // padded tokens
constexpr int NWIN  = 2048;    // B * 64 windows
constexpr int ROWS  = NWIN * NTOK;   // 100352
constexpr int HSZ   = 56;
constexpr int SHIFT = 3;
constexpr float SCALE = 0.17677669529663687f; // 32^-0.5

typedef _Float16 v16h  __attribute__((ext_vector_type(16)));
typedef _Float16 half8 __attribute__((ext_vector_type(8)));
typedef float    v8f   __attribute__((ext_vector_type(8)));

__device__ __forceinline__ v8f v8f_zero() {
  v8f z = {0.f,0.f,0.f,0.f,0.f,0.f,0.f,0.f};
  return z;
}

__device__ __forceinline__ v8f wmma16(v16h a, v16h b, v8f c) {
  return __builtin_amdgcn_wmma_f32_16x16x32_f16(false, a, false, b, (short)0, c, false, false);
}

// A-fragment (16x32 f16). Caller passes p = rowBase + kb + g*16.
// slots 0..15 -> K = kb + g*16 + 0..15 (contiguous run)
__device__ __forceinline__ v16h frag_a(const _Float16* p) {
  half8 lo = *reinterpret_cast<const half8*>(p);
  half8 hi = *reinterpret_cast<const half8*>(p + 8);
  v16h r;
#pragma unroll
  for (int i = 0; i < 8; ++i) { r[i] = lo[i]; r[i + 8] = hi[i]; }
  return r;
}

// B-fragment (32x16 f16). Caller passes p = colBase + kb + g*16.
// slots 0..15 -> K = kb + g*16 + 0..15 (contiguous run)
__device__ __forceinline__ v16h frag_b(const _Float16* p) {
  half8 lo = *reinterpret_cast<const half8*>(p);
  half8 hi = *reinterpret_cast<const half8*>(p + 8);
  v16h r;
#pragma unroll
  for (int i = 0; i < 8; ++i) { r[i] = lo[i]; r[i + 8] = hi[i]; }
  return r;
}

__device__ __forceinline__ int reg3(int s) { return s < 49 ? 0 : (s < 53 ? 1 : 2); }

// ---------------- LN1 + cyclic shift + window partition -> f16 ----------------
__global__ void k_ln_shift_window(const float* __restrict__ x,
                                  const float* __restrict__ gam,
                                  const float* __restrict__ bet,
                                  _Float16* __restrict__ out) {
  int row  = blockIdx.x * 8 + (threadIdx.x >> 5);
  int lane = threadIdx.x & 31;
  int win = row / NTOK, t = row - win * NTOK;
  int b = win >> 6, wh = (win >> 3) & 7, ww = win & 7;
  int r = t / 7, c = t - r * 7;
  int h0 = (wh * 7 + r + SHIFT) % HSZ;
  int w0 = (ww * 7 + c + SHIFT) % HSZ;
  const float* src = x + ((size_t)(b * 3136 + h0 * HSZ + w0)) * CDIM + lane * 8;
  float v[8];
  float4 p0 = *reinterpret_cast<const float4*>(src);
  float4 p1 = *reinterpret_cast<const float4*>(src + 4);
  v[0]=p0.x; v[1]=p0.y; v[2]=p0.z; v[3]=p0.w;
  v[4]=p1.x; v[5]=p1.y; v[6]=p1.z; v[7]=p1.w;
  float s = 0.f, s2 = 0.f;
#pragma unroll
  for (int i = 0; i < 8; ++i) { s += v[i]; s2 += v[i] * v[i]; }
#pragma unroll
  for (int off = 16; off > 0; off >>= 1) { s += __shfl_xor(s, off); s2 += __shfl_xor(s2, off); }
  float mu  = s  * (1.f / 256.f);
  float inv = rsqrtf(s2 * (1.f / 256.f) - mu * mu + 1e-5f);
  half8 o;
#pragma unroll
  for (int i = 0; i < 8; ++i) {
    int ch = lane * 8 + i;
    o[i] = (_Float16)((v[i] - mu) * inv * gam[ch] + bet[ch]);
  }
  *reinterpret_cast<half8*>(out + (size_t)row * CDIM + lane * 8) = o;
}

// ---------------- plain LN (rows in place order) -> f16 ----------------
__global__ void k_ln(const float* __restrict__ x,
                     const float* __restrict__ gam,
                     const float* __restrict__ bet,
                     _Float16* __restrict__ out) {
  int row  = blockIdx.x * 8 + (threadIdx.x >> 5);
  int lane = threadIdx.x & 31;
  const float* src = x + (size_t)row * CDIM + lane * 8;
  float v[8];
  float4 p0 = *reinterpret_cast<const float4*>(src);
  float4 p1 = *reinterpret_cast<const float4*>(src + 4);
  v[0]=p0.x; v[1]=p0.y; v[2]=p0.z; v[3]=p0.w;
  v[4]=p1.x; v[5]=p1.y; v[6]=p1.z; v[7]=p1.w;
  float s = 0.f, s2 = 0.f;
#pragma unroll
  for (int i = 0; i < 8; ++i) { s += v[i]; s2 += v[i] * v[i]; }
#pragma unroll
  for (int off = 16; off > 0; off >>= 1) { s += __shfl_xor(s, off); s2 += __shfl_xor(s2, off); }
  float mu  = s  * (1.f / 256.f);
  float inv = rsqrtf(s2 * (1.f / 256.f) - mu * mu + 1e-5f);
  half8 o;
#pragma unroll
  for (int i = 0; i < 8; ++i) {
    int ch = lane * 8 + i;
    o[i] = (_Float16)((v[i] - mu) * inv * gam[ch] + bet[ch]);
  }
  *reinterpret_cast<half8*>(out + (size_t)row * CDIM + lane * 8) = o;
}

// ---------------- weight convert f32 (K,N) -> f16 transposed (N,K) ----------------
__global__ void k_wcvt(const float* __restrict__ src, _Float16* __restrict__ dst,
                       int K, int N) {
  int idx = blockIdx.x * 256 + threadIdx.x;
  if (idx >= K * N) return;
  int k = idx / N, n = idx - k * N;
  dst[(size_t)n * K + k] = (_Float16)src[idx];
}

// ---------------- relative position bias table -> (heads,49,49) f32 ----------------
__global__ void k_relbias(const float* __restrict__ tab, float* __restrict__ relb) {
  int idx = blockIdx.x * 256 + threadIdx.x;
  if (idx >= HEADS * NTOK * NTOK) return;
  int head = idx / (NTOK * NTOK);
  int ij = idx - head * (NTOK * NTOK);
  int i = ij / NTOK, j = ij - i * NTOK;
  int yi = i / 7, xi = i - yi * 7, yj = j / 7, xj = j - yj * 7;
  int ridx = (yi - yj + 6) * 13 + (xi - xj + 6);
  relb[idx] = tab[ridx * HEADS + head];
}

__global__ void k_zero16(uint4* __restrict__ p, int n16) {
  int i = blockIdx.x * 256 + threadIdx.x;
  if (i < n16) p[i] = make_uint4(0u, 0u, 0u, 0u);
}

// ---------------- generic 16x64 output tile, A row-major MxK, Bt NxK ----------------
template <int KDIM>
__device__ __forceinline__ void gemm_tile16x64(const _Float16* __restrict__ A,
                                               const _Float16* __restrict__ Bt,
                                               int mtile, int n0, int lane, v8f acc[4]) {
  int g = lane >> 4;
  const _Float16* aRow  = A  + (size_t)(mtile * 16 + (lane & 15)) * KDIM;
  const _Float16* bRow0 = Bt + (size_t)(n0 + (lane & 15)) * KDIM;
#pragma unroll
  for (int ks = 0; ks < KDIM / 32; ++ks) {
    int kb = ks * 32;
    __builtin_prefetch(aRow + kb + 64, 0, 0);
    v16h a = frag_a(aRow + kb + g * 16);
#pragma unroll
    for (int j = 0; j < 4; ++j) {
      v16h b = frag_b(bRow0 + (size_t)j * 16 * KDIM + kb + g * 16);
      acc[j] = wmma16(a, b, acc[j]);
    }
  }
}

// ---------------- QKV GEMM: (100352,256)f16 @ (768,256)^T + scatter to q/k/vT ----
__global__ void k_gemm_qkv(const _Float16* __restrict__ A, const _Float16* __restrict__ Bt,
                           const float* __restrict__ bias,
                           _Float16* __restrict__ qb, _Float16* __restrict__ kb2,
                           _Float16* __restrict__ vT) {
  int wave = threadIdx.x >> 5, lane = threadIdx.x & 31, g = lane >> 4;
  int gid = blockIdx.x * 8 + wave;
  int mtile = gid / 12, nch = gid - mtile * 12;
  v8f acc[4] = {v8f_zero(), v8f_zero(), v8f_zero(), v8f_zero()};
  gemm_tile16x64<CDIM>(A, Bt, mtile, nch * 64, lane, acc);
#pragma unroll
  for (int j = 0; j < 4; ++j) {
    int nn = nch * 64 + j * 16 + (lane & 15);
    float bval = bias[nn];
    int sgrp = nn >> 8, rem = nn & 255, head = rem >> 5, d = rem & 31;
#pragma unroll
    for (int r = 0; r < 8; ++r) {
      int m = mtile * 16 + r + 8 * g;
      int win = m / NTOK, t = m - win * NTOK;
      _Float16 hv = (_Float16)(acc[j][r] + bval);
      size_t base = (size_t)win * HEADS + head;
      if (sgrp == 0)      qb [(base * NPAD + t) * HD + d] = hv;
      else if (sgrp == 1) kb2[(base * NPAD + t) * HD + d] = hv;
      else                vT [(base * HD + d) * NPAD + t] = hv;
    }
  }
}

// ---------------- fused window attention: 1 block per window, 1 wave per head ----
__global__ void k_attn(const _Float16* __restrict__ qb, const _Float16* __restrict__ kb2,
                       const _Float16* __restrict__ vT, const float* __restrict__ relb,
                       _Float16* __restrict__ out) {
  extern __shared__ _Float16 smem[];               // 8 waves * 64 * 72 halfs
  int head = threadIdx.x >> 5, lane = threadIdx.x & 31;
  int g = lane >> 4, ln15 = lane & 15;
  int win = blockIdx.x;
  int wh = (win >> 3) & 7, ww = win & 7;
  _Float16* pw = smem + head * (NPAD * 72);
  const _Float16* qbase = qb  + ((size_t)win * HEADS + head) * NPAD * HD;
  const _Float16* kbase = kb2 + ((size_t)win * HEADS + head) * NPAD * HD;
  const _Float16* vbase = vT  + ((size_t)win * HEADS + head) * HD * NPAD;

  v16h aq[4], bk[4];
#pragma unroll
  for (int mt = 0; mt < 4; ++mt) aq[mt] = frag_a(qbase + (mt * 16 + ln15) * HD + g * 16);
#pragma unroll
  for (int nt = 0; nt < 4; ++nt) bk[nt] = frag_b(kbase + (nt * 16 + ln15) * HD + g * 16);

  v8f s[4][4];
#pragma unroll
  for (int mt = 0; mt < 4; ++mt)
#pragma unroll
    for (int nt = 0; nt < 4; ++nt)
      s[mt][nt] = wmma16(aq[mt], bk[nt], v8f_zero());

  // scale + relative-position bias + shifted-window mask (computed inline)
  const float* rb = relb + head * (NTOK * NTOK);
#pragma unroll
  for (int mt = 0; mt < 4; ++mt)
#pragma unroll
    for (int nt = 0; nt < 4; ++nt) {
      int n = nt * 16 + ln15;
#pragma unroll
      for (int r = 0; r < 8; ++r) {
        int m = mt * 16 + r + 8 * g;
        float v = s[mt][nt][r] * SCALE;
        if (n >= NTOK) {
          v = -1e30f;                      // padded keys: exact 0 after softmax
        } else if (m < NTOK) {
          v += rb[m * NTOK + n];
          int rm = m / 7, cm = m - rm * 7, rn = n / 7, cn = n - rn * 7;
          int cm9 = 3 * reg3(wh * 7 + rm) + reg3(ww * 7 + cm);
          int cn9 = 3 * reg3(wh * 7 + rn) + reg3(ww * 7 + cn);
          if (cm9 != cn9) v -= 100.f;
        }
        s[mt][nt][r] = v;
      }
    }

  // row softmax (rows live across 16 lanes of each half-wave) -> P (f16) in LDS
#pragma unroll
  for (int mt = 0; mt < 4; ++mt) {
#pragma unroll
    for (int r = 0; r < 8; ++r) {
      float mx = fmaxf(fmaxf(s[mt][0][r], s[mt][1][r]), fmaxf(s[mt][2][r], s[mt][3][r]));
#pragma unroll
      for (int off = 1; off < 16; off <<= 1) mx = fmaxf(mx, __shfl_xor(mx, off));
      float sum = 0.f;
#pragma unroll
      for (int nt = 0; nt < 4; ++nt) {
        float e = __expf(s[mt][nt][r] - mx);
        s[mt][nt][r] = e;
        sum += e;
      }
#pragma unroll
      for (int off = 1; off < 16; off <<= 1) sum += __shfl_xor(sum, off);
      float inv = 1.0f / sum;
      int m = mt * 16 + r + 8 * g;
#pragma unroll
      for (int nt = 0; nt < 4; ++nt) {
        int n = nt * 16 + ln15;
        pw[m * 72 + n] = (_Float16)(s[mt][nt][r] * inv);
      }
    }
  }
  __syncthreads();

  // O = P (64x64) @ V (64x32); V rows >= 49 are zero-padded
  v16h bv[2][2];
#pragma unroll
  for (int kk = 0; kk < 2; ++kk)
#pragma unroll
    for (int nt = 0; nt < 2; ++nt)
      bv[kk][nt] = frag_b(vbase + (nt * 16 + ln15) * NPAD + kk * 32 + g * 16);

  v8f o[4][2];
#pragma unroll
  for (int mt = 0; mt < 4; ++mt) { o[mt][0] = v8f_zero(); o[mt][1] = v8f_zero(); }
#pragma unroll
  for (int mt = 0; mt < 4; ++mt)
#pragma unroll
    for (int kk = 0; kk < 2; ++kk) {
      v16h pa = frag_a(pw + (mt * 16 + ln15) * 72 + kk * 32 + g * 16);
#pragma unroll
      for (int nt = 0; nt < 2; ++nt) o[mt][nt] = wmma16(pa, bv[kk][nt], o[mt][nt]);
    }

#pragma unroll
  for (int mt = 0; mt < 4; ++mt)
#pragma unroll
    for (int nt = 0; nt < 2; ++nt)
#pragma unroll
      for (int r = 0; r < 8; ++r) {
        int m = mt * 16 + r + 8 * g;
        if (m < NTOK)
          out[((size_t)win * NTOK + m) * CDIM + head * HD + nt * 16 + ln15] =
              (_Float16)o[mt][nt][r];
      }
}

// ---------------- proj GEMM + window-reverse + un-shift + residual -> d_out ------
__global__ void k_gemm_proj(const _Float16* __restrict__ A, const _Float16* __restrict__ Bt,
                            const float* __restrict__ bias, const float* __restrict__ x,
                            float* __restrict__ out) {
  int wave = threadIdx.x >> 5, lane = threadIdx.x & 31, g = lane >> 4;
  int gid = blockIdx.x * 8 + wave;
  int mtile = gid >> 2, nch = gid & 3;
  v8f acc[4] = {v8f_zero(), v8f_zero(), v8f_zero(), v8f_zero()};
  gemm_tile16x64<CDIM>(A, Bt, mtile, nch * 64, lane, acc);
#pragma unroll
  for (int j = 0; j < 4; ++j) {
    int n = nch * 64 + j * 16 + (lane & 15);
    float bb = bias[n];
#pragma unroll
    for (int r = 0; r < 8; ++r) {
      int m = mtile * 16 + r + 8 * g;
      int win = m / NTOK, t = m - win * NTOK;
      int b = win >> 6, wh = (win >> 3) & 7, ww = win & 7;
      int rr = t / 7, cc = t - rr * 7;
      int h0 = (wh * 7 + rr + SHIFT) % HSZ;
      int w0 = (ww * 7 + cc + SHIFT) % HSZ;
      size_t idx = ((size_t)(b * 3136 + h0 * HSZ + w0)) * CDIM + n;
      out[idx] = x[idx] + acc[j][r] + bb;
    }
  }
}

// ---------------- fused MLP: FC1 + GELU + FC2 + residual, hidden kept in LDS -----
__global__ void k_mlp(const _Float16* __restrict__ xn, const _Float16* __restrict__ w1T,
                      const float* __restrict__ b1, const _Float16* __restrict__ w2T,
                      const float* __restrict__ b2, float* __restrict__ io) {
  extern __shared__ _Float16 smem[];
  _Float16* sA = smem;              // 64 x 264 (padded stride)
  _Float16* sH = smem + 64 * 264;   // 64 x 1032 (padded stride)
  int tid = threadIdx.x, wave = tid >> 5, lane = tid & 31;
  int g = lane >> 4, ln15 = lane & 15;
  size_t row0 = (size_t)blockIdx.x * 64;

  // stage the 64x256 input strip into LDS
  for (int i = tid; i < 64 * 32; i += 256) {
    int rr = i >> 5, ch = i & 31;
    *reinterpret_cast<half8*>(sA + rr * 264 + ch * 8) =
        *reinterpret_cast<const half8*>(xn + (row0 + rr) * CDIM + ch * 8);
  }
  __syncthreads();

  // FC1 (K=256, N=1024) + bias + exact GELU -> sH
  for (int gnt = 0; gnt < 8; ++gnt) {
    int ncol = (wave * 8 + gnt) * 16 + ln15;
    v8f acc[4] = {v8f_zero(), v8f_zero(), v8f_zero(), v8f_zero()};
    const _Float16* bRow = w1T + (size_t)ncol * 256;
#pragma unroll
    for (int ks = 0; ks < 8; ++ks) {
      v16h b = frag_b(bRow + ks * 32 + g * 16);
#pragma unroll
      for (int mt = 0; mt < 4; ++mt) {
        v16h a = frag_a(sA + (mt * 16 + ln15) * 264 + ks * 32 + g * 16);
        acc[mt] = wmma16(a, b, acc[mt]);
      }
    }
    float bb = b1[ncol];
#pragma unroll
    for (int mt = 0; mt < 4; ++mt)
#pragma unroll
      for (int r = 0; r < 8; ++r) {
        int m = mt * 16 + r + 8 * g;
        float v = acc[mt][r] + bb;
        v = 0.5f * v * (1.f + erff(v * 0.7071067811865475f));
        sH[m * 1032 + ncol] = (_Float16)v;
      }
  }
  __syncthreads();

  // FC2 (K=1024, N=256) + bias + residual from io
  for (int gnt = 0; gnt < 2; ++gnt) {
    int ncol = (wave * 2 + gnt) * 16 + ln15;
    v8f acc[4] = {v8f_zero(), v8f_zero(), v8f_zero(), v8f_zero()};
    const _Float16* bRow = w2T + (size_t)ncol * 1024;
#pragma unroll
    for (int ks = 0; ks < 32; ++ks) {
      v16h b = frag_b(bRow + ks * 32 + g * 16);
#pragma unroll
      for (int mt = 0; mt < 4; ++mt) {
        v16h a = frag_a(sH + (mt * 16 + ln15) * 1032 + ks * 32 + g * 16);
        acc[mt] = wmma16(a, b, acc[mt]);
      }
    }
    float bb = b2[ncol];
#pragma unroll
    for (int mt = 0; mt < 4; ++mt)
#pragma unroll
      for (int r = 0; r < 8; ++r) {
        int m = mt * 16 + r + 8 * g;
        size_t idx = (row0 + m) * CDIM + ncol;
        io[idx] = io[idx] + acc[mt][r] + bb;
      }
  }
}

// ---------------- host-side orchestration ----------------
extern "C" void kernel_launch(void* const* d_in, const int* in_sizes, int n_in,
                              void* d_out, int out_size, void* d_ws, size_t ws_size,
                              hipStream_t stream) {
  (void)in_sizes; (void)n_in; (void)out_size; (void)ws_size;
  const float* x      = (const float*)d_in[0];
  const float* n1g    = (const float*)d_in[1];
  const float* n1b    = (const float*)d_in[2];
  const float* qkv_w  = (const float*)d_in[3];
  const float* qkv_b  = (const float*)d_in[4];
  const float* btab   = (const float*)d_in[5];
  const float* proj_w = (const float*)d_in[6];
  const float* proj_b = (const float*)d_in[7];
  const float* n2g    = (const float*)d_in[8];
  const float* n2b    = (const float*)d_in[9];
  const float* fc1_w  = (const float*)d_in[10];
  const float* fc1_b  = (const float*)d_in[11];
  const float* fc2_w  = (const float*)d_in[12];
  const float* fc2_b  = (const float*)d_in[13];
  float* out = (float*)d_out;

  char* ws = (char*)d_ws;
  size_t off = 0;
  auto alloc = [&](size_t bytes) {
    char* p = ws + off;
    off = (off + bytes + 255) & ~(size_t)255;
    return p;
  };
  _Float16* win_a  = (_Float16*)alloc((size_t)ROWS * CDIM * 2);
  _Float16* qkvT   = (_Float16*)alloc((size_t)768 * 256 * 2);
  _Float16* projT  = (_Float16*)alloc((size_t)256 * 256 * 2);
  _Float16* fc1T   = (_Float16*)alloc((size_t)1024 * 256 * 2);
  _Float16* fc2T   = (_Float16*)alloc((size_t)256 * 1024 * 2);
  float*    relb   = (float*)alloc((size_t)HEADS * NTOK * NTOK * 4);
  _Float16* qbuf   = (_Float16*)alloc((size_t)NWIN * HEADS * NPAD * HD * 2);
  _Float16* kbuf   = (_Float16*)alloc((size_t)NWIN * HEADS * NPAD * HD * 2);
  _Float16* vTb    = (_Float16*)alloc((size_t)NWIN * HEADS * HD * NPAD * 2);
  _Float16* attn_o = (_Float16*)alloc((size_t)ROWS * CDIM * 2);
  _Float16* xn     = (_Float16*)alloc((size_t)ROWS * CDIM * 2);

  // weight prep (small)
  k_wcvt<<<(256 * 768 + 255) / 256, 256, 0, stream>>>(qkv_w, qkvT, 256, 768);
  k_wcvt<<<(256 * 256 + 255) / 256, 256, 0, stream>>>(proj_w, projT, 256, 256);
  k_wcvt<<<(256 * 1024 + 255) / 256, 256, 0, stream>>>(fc1_w, fc1T, 256, 1024);
  k_wcvt<<<(1024 * 256 + 255) / 256, 256, 0, stream>>>(fc2_w, fc2T, 1024, 256);
  k_relbias<<<(HEADS * NTOK * NTOK + 255) / 256, 256, 0, stream>>>(btab, relb);
  int vbytes16 = (NWIN * HEADS * HD * NPAD * 2) / 16;
  k_zero16<<<(vbytes16 + 255) / 256, 256, 0, stream>>>((uint4*)vTb, vbytes16);

  // main pipeline
  k_ln_shift_window<<<ROWS / 8, 256, 0, stream>>>(x, n1g, n1b, win_a);
  k_gemm_qkv<<<(6272 * 12) / 8, 256, 0, stream>>>(win_a, qkvT, qkv_b, qbuf, kbuf, vTb);
  k_attn<<<NWIN, 256, (size_t)HEADS * NPAD * 72 * 2, stream>>>(qbuf, kbuf, vTb, relb, attn_o);
  k_gemm_proj<<<(6272 * 4) / 8, 256, 0, stream>>>(attn_o, projT, proj_b, x, out);
  k_ln<<<ROWS / 8, 256, 0, stream>>>(out, n2g, n2b, xn);
  k_mlp<<<ROWS / 64, 256, (size_t)(64 * 264 + 64 * 1032) * 2, stream>>>(
      xn, fc1T, fc1_b, fc2T, fc2_b, out);
}
